// RetrocausalAttention_28939489640882
// MI455X (gfx1250) — matrix operand order repaired and checked by reading, not compiled, not run on hardware
//
#include <hip/hip_runtime.h>
#include <hip/hip_bf16.h>
#include <cstdint>

#define BQ 2
#define SQ 2048
#define DQ 1024
#define HQ 16
#define HDQ 64
#define MQ (BQ * SQ)        // 4096 rows
#define LN_EPS 1e-5f
#define LDS_STR 40          // padded K-stride in halves (80B) to spread LDS banks

typedef __attribute__((ext_vector_type(16))) __bf16 v16bf;
typedef __attribute__((ext_vector_type(8)))  float  v8f;

// ---------------------------------------------------------------------------
// WMMA helpers (CDNA5 wave32, 16x16x32 bf16 -> f32)
// ---------------------------------------------------------------------------
__device__ inline v8f v8f_zero() {
    v8f z;
#pragma unroll
    for (int i = 0; i < 8; ++i) z[i] = 0.0f;
    return z;
}

__device__ inline v8f wmma_bf16(v16bf a, v16bf b, v8f c) {
    return __builtin_amdgcn_wmma_f32_16x16x32_bf16(false, a, false, b,
                                                   (short)0, c, false, false);
}

// A-matrix fragment 16x32 bf16 (ISA 7.12.2):
//  lane<16 : row=lane,    halves 0..7 = K 0..7,  8..15 = K 16..23
//  lane>=16: row=lane-16, halves 0..7 = K 8..15, 8..15 = K 24..31
__device__ inline v16bf load_afrag(const __bf16* base, int stride, int lane) {
    const int r   = lane & 15;
    const int klo = (lane & 16) ? 8 : 0;
    const __bf16* p = base + (size_t)r * stride + klo;
    union { v16bf v; uint4 q[2]; } u;
    u.q[0] = *(const uint4*)(p);
    u.q[1] = *(const uint4*)(p + 16);
    return u.v;
}

// B-matrix fragment 32x16 bf16: lane holds column N=lane&15;
//  lanes 0..15: K 0..7,16..23, lanes 16..31: K 8..15,24..31. base is N-major (K contiguous).
__device__ inline v16bf load_bfrag(const __bf16* base, int stride, int lane) {
    const int n   = lane & 15;
    const int klo = (lane & 16) ? 8 : 0;
    const __bf16* p = base + (size_t)n * stride + klo;
    union { v16bf v; uint4 q[2]; } u;
    u.q[0] = *(const uint4*)(p);
    u.q[1] = *(const uint4*)(p + 16);
    return u.v;
}

// ---------------------------------------------------------------------------
// f32 -> bf16 (straight)
// ---------------------------------------------------------------------------
__global__ void k_cvt_bf16(const float* __restrict__ in, __bf16* __restrict__ out, int n) {
    int i = blockIdx.x * blockDim.x + threadIdx.x;
    if (i < n) out[i] = (__bf16)in[i];
}

// f32 [R,C] -> bf16 [C,R] (transpose-convert for weights), 32x32 tiles
__global__ __launch_bounds__(256) void k_cvt_T(const float* __restrict__ in,
                                               __bf16* __restrict__ out, int R, int C) {
    __shared__ __bf16 t[32][33];
    const int c0 = blockIdx.x * 32, r0 = blockIdx.y * 32;
    const int tx = threadIdx.x & 31, ty = threadIdx.x >> 5;
#pragma unroll
    for (int i = 0; i < 4; ++i) {
        const int r = ty + i * 8;
        t[tx][r] = (__bf16)in[(size_t)(r0 + r) * C + c0 + tx];
    }
    __syncthreads();
#pragma unroll
    for (int i = 0; i < 4; ++i) {
        const int c = ty + i * 8;
        out[(size_t)(c0 + c) * R + r0 + tx] = t[c][tx];
    }
}

// V [B*S, D] (head-sliced) -> Vt [B*H, HD, S]
__global__ __launch_bounds__(256) void k_transpose_v(const __bf16* __restrict__ V,
                                                     __bf16* __restrict__ Vt) {
    __shared__ __bf16 ts[64][33];
    const int bh = blockIdx.y;
    const int b = bh >> 4, h = bh & 15;
    const int s0 = blockIdx.x * 32;
    const int tid = threadIdx.x;
    {
        const int d = tid & 63, si0 = tid >> 6;     // 4 s-rows per pass
#pragma unroll
        for (int j = 0; j < 8; ++j) {
            const int si = si0 + j * 4;
            ts[d][si] = V[((size_t)b * SQ + s0 + si) * DQ + h * HDQ + d];
        }
    }
    __syncthreads();
    {
        const int si = tid & 31, dj = tid >> 5;     // 8 d-rows per pass
#pragma unroll
        for (int j = 0; j < 8; ++j) {
            const int d = dj + j * 8;
            Vt[((size_t)bh * HDQ + d) * SQ + s0 + si] = ts[d][si];
        }
    }
}

// ---------------------------------------------------------------------------
// GEMM: C[M,N] = A[M,K](bf16) @ Wt[N,K](bf16, pre-transposed) + bias
// 128x128 tile, BK=32, 8 waves (4M x 2N), double-buffered LDS, prefetch +2.
// ---------------------------------------------------------------------------
template <typename OUT>
__global__ __launch_bounds__(256) void gemm_wmma(
    const __bf16* __restrict__ A, const __bf16* __restrict__ Wt,
    const float* __restrict__ bias, OUT* __restrict__ C,
    int M, int N, int K)
{
    __shared__ __align__(16) __bf16 As[2][128 * LDS_STR];
    __shared__ __align__(16) __bf16 Ws[2][128 * LDS_STR];

    const int tid  = threadIdx.x;
    const int lane = tid & 31;
    const int wave = tid >> 5;
    const int wm   = wave & 3;    // 4 waves along M (32 rows each)
    const int wn   = wave >> 2;   // 2 waves along N (64 cols each)
    const int m0 = blockIdx.y * 128;
    const int n0 = blockIdx.x * 128;

    const int row = tid >> 1;           // 0..127
    const int seg = (tid & 1) * 16;     // 0 / 16
    const __bf16* gA = A  + (size_t)(m0 + row) * K + seg;
    const __bf16* gW = Wt + (size_t)(n0 + row) * K + seg;

    v8f acc[2][4];
#pragma unroll
    for (int i = 0; i < 2; ++i)
#pragma unroll
        for (int j = 0; j < 4; ++j) acc[i][j] = v8f_zero();

    const int nk = K >> 5;

    // prologue: tile 0 -> buf 0
    {
        uint4 a0 = *(const uint4*)(gA);
        uint4 a1 = *(const uint4*)(gA + 8);
        uint4 w0 = *(const uint4*)(gW);
        uint4 w1 = *(const uint4*)(gW + 8);
        *(uint4*)&As[0][row * LDS_STR + seg]     = a0;
        *(uint4*)&As[0][row * LDS_STR + seg + 8] = a1;
        *(uint4*)&Ws[0][row * LDS_STR + seg]     = w0;
        *(uint4*)&Ws[0][row * LDS_STR + seg + 8] = w1;
    }

    for (int kt = 0; kt < nk; ++kt) {
        __syncthreads();
        // issue next-tile global loads early (overlap with WMMA below)
        uint4 a0, a1, w0, w1;
        const bool more = (kt + 1 < nk);
        if (more) {
            const int k1 = (kt + 1) << 5;
            a0 = *(const uint4*)(gA + k1);
            a1 = *(const uint4*)(gA + k1 + 8);
            w0 = *(const uint4*)(gW + k1);
            w1 = *(const uint4*)(gW + k1 + 8);
            if (kt + 2 < nk) {
                __builtin_prefetch(gA + ((kt + 2) << 5), 0, 1);
                __builtin_prefetch(gW + ((kt + 2) << 5), 0, 1);
            }
        }

        const __bf16* as = &As[kt & 1][0];
        const __bf16* ws = &Ws[kt & 1][0];
        v16bf af[2], bfr[4];
#pragma unroll
        for (int i = 0; i < 2; ++i)
            af[i] = load_afrag(&as[(wm * 32 + i * 16) * LDS_STR], LDS_STR, lane);
#pragma unroll
        for (int j = 0; j < 4; ++j)
            bfr[j] = load_bfrag(&ws[(wn * 64 + j * 16) * LDS_STR], LDS_STR, lane);
#pragma unroll
        for (int i = 0; i < 2; ++i)
#pragma unroll
            for (int j = 0; j < 4; ++j)
                acc[i][j] = wmma_bf16(af[i], bfr[j], acc[i][j]);

        if (more) {
            const int nb = (kt + 1) & 1;
            *(uint4*)&As[nb][row * LDS_STR + seg]     = a0;
            *(uint4*)&As[nb][row * LDS_STR + seg + 8] = a1;
            *(uint4*)&Ws[nb][row * LDS_STR + seg]     = w0;
            *(uint4*)&Ws[nb][row * LDS_STR + seg + 8] = w1;
        }
    }

    // C layout: VGPR v, lane L: M = v + (L<16?0:8), N = L&15
    const int nn = lane & 15;
    const int rb = (lane & 16) ? 8 : 0;
#pragma unroll
    for (int i = 0; i < 2; ++i)
#pragma unroll
        for (int j = 0; j < 4; ++j) {
            const int col = n0 + wn * 64 + j * 16 + nn;
            const float bs = bias[col];
#pragma unroll
            for (int v = 0; v < 8; ++v) {
                const int row2 = m0 + wm * 32 + i * 16 + rb + v;
                C[(size_t)row2 * N + col] = (OUT)(acc[i][j][v] + bs);
            }
        }
}

// ---------------------------------------------------------------------------
// Causal flash attention. Barrier-free: each wave owns 16 q rows and loops
// only over its own causal kv range. K frags straight from global (row-major),
// V frags straight from global Vt (pre-transposed), P via per-wave LDS tile.
// ---------------------------------------------------------------------------
__global__ __launch_bounds__(256) void fwd_attn(
    const __bf16* __restrict__ Q, const __bf16* __restrict__ K,
    const __bf16* __restrict__ Vt,                       // [B*H, HD, S]
    float* __restrict__ fwdF, __bf16* __restrict__ comb)
{
    __shared__ __align__(16) __bf16 Ps[8][16 * LDS_STR];

    const int tid  = threadIdx.x;
    const int lane = tid & 31;
    const int wave = tid >> 5;
    const int qb   = blockIdx.x * 128;
    const int bh   = blockIdx.y;
    const int b = bh >> 4, h = bh & 15;
    const size_t baseQK = ((size_t)b * SQ) * DQ + (size_t)h * HDQ;  // stride DQ
    const size_t baseV  = (size_t)bh * HDQ * SQ;                    // stride SQ

    const int qrow0 = qb + wave * 16;
    const v16bf aq0 = load_afrag(Q + baseQK + (size_t)qrow0 * DQ,      DQ, lane);
    const v16bf aq1 = load_afrag(Q + baseQK + (size_t)qrow0 * DQ + 32, DQ, lane);

    v8f O[4];
#pragma unroll
    for (int d = 0; d < 4; ++d) O[d] = v8f_zero();
    float mrow[8], lrow[8];
#pragma unroll
    for (int v = 0; v < 8; ++v) { mrow[v] = -1e30f; lrow[v] = 0.0f; }

    const int rb   = (lane & 16) ? 8 : 0;
    const int ncol = lane & 15;
    const float scale = 0.125f;  // 1/sqrt(64)
    __bf16* P = &Ps[wave][0];

    for (int kc = 0; kc < qrow0 + 16; kc += 32) {
        const __bf16* kp = K + baseQK + (size_t)kc * DQ;
        const v16bf bk00 = load_bfrag(kp,                DQ, lane);
        const v16bf bk01 = load_bfrag(kp + 32,           DQ, lane);
        const v16bf bk10 = load_bfrag(kp + 16 * DQ,      DQ, lane);
        const v16bf bk11 = load_bfrag(kp + 16 * DQ + 32, DQ, lane);
        v8f s0 = v8f_zero(), s1 = v8f_zero();
        s0 = wmma_bf16(aq0, bk00, s0);
        s0 = wmma_bf16(aq1, bk01, s0);
        s1 = wmma_bf16(aq0, bk10, s1);
        s1 = wmma_bf16(aq1, bk11, s1);

        const int col0 = kc + ncol;
        const int col1 = col0 + 16;
#pragma unroll
        for (int v = 0; v < 8; ++v) {
            const int row = qrow0 + rb + v;
            float a = s0[v] * scale; if (col0 > row) a = -1e30f;
            float c = s1[v] * scale; if (col1 > row) c = -1e30f;
            float mx = fmaxf(a, c);
#pragma unroll
            for (int off = 1; off < 16; off <<= 1)
                mx = fmaxf(mx, __shfl_xor(mx, off, 32));
            const float mn = fmaxf(mrow[v], mx);
            const float sc = __expf(mrow[v] - mn);
            const float p0 = __expf(a - mn);
            const float p1 = __expf(c - mn);
            float ps = p0 + p1;
#pragma unroll
            for (int off = 1; off < 16; off <<= 1)
                ps += __shfl_xor(ps, off, 32);
            lrow[v] = lrow[v] * sc + ps;
            mrow[v] = mn;
#pragma unroll
            for (int d = 0; d < 4; ++d) O[d][v] *= sc;
            P[(rb + v) * LDS_STR + ncol]      = (__bf16)p0;
            P[(rb + v) * LDS_STR + 16 + ncol] = (__bf16)p1;
        }
        // PV: A = P (16x32), B = Vt columns (K=kv contiguous in global)
        const v16bf ap = load_afrag(P, LDS_STR, lane);
#pragma unroll
        for (int d = 0; d < 4; ++d) {
            const v16bf bv = load_bfrag(Vt + baseV + (size_t)(d * 16) * SQ + kc, SQ, lane);
            O[d] = wmma_bf16(ap, bv, O[d]);
        }
    }

#pragma unroll
    for (int d = 0; d < 4; ++d)
#pragma unroll
        for (int v = 0; v < 8; ++v) {
            const int row = qrow0 + rb + v;
            const float o = O[d][v] / lrow[v];
            const size_t rr = (size_t)b * SQ + row;
            fwdF[rr * DQ + h * HDQ + d * 16 + ncol] = o;
            comb[rr * (2 * DQ) + h * HDQ + d * 16 + ncol] = (__bf16)o;
        }
}

// ---------------------------------------------------------------------------
// Backward window attention (WIN=32 future). One warp per (b,h,q) row.
// Last row (q==S-1): uniform softmax over ALL S positions (reference corner).
// ---------------------------------------------------------------------------
__global__ __launch_bounds__(256) void bwd_attn(
    const __bf16* __restrict__ Qb, const __bf16* __restrict__ Kb,
    const __bf16* __restrict__ Vb, float* __restrict__ bwdF,
    __bf16* __restrict__ comb)
{
    const int lane = threadIdx.x & 31;
    const int warp = threadIdx.x >> 5;
    const int row  = blockIdx.x * 8 + warp;   // 0 .. B*H*S-1
    const int q  = row & (SQ - 1);
    const int bh = row >> 11;
    const int h  = bh & (HQ - 1);
    const int b  = bh >> 4;
    const size_t base = ((size_t)b * SQ) * DQ + (size_t)h * HDQ;
    const int d0 = lane * 2;

    float o0 = 0.0f, o1 = 0.0f;
    if (q == SQ - 1) {
        for (int kv = 0; kv < SQ; ++kv) {
            union { unsigned u; __bf16 e[2]; } uv;
            uv.u = *(const unsigned*)(Vb + base + (size_t)kv * DQ + d0);
            o0 += (float)uv.e[0];
            o1 += (float)uv.e[1];
        }
        o0 *= (1.0f / SQ);
        o1 *= (1.0f / SQ);
    } else {
        const int nkv = (32 < SQ - 1 - q) ? 32 : (SQ - 1 - q);
        float s = -1e30f;
        if (lane < nkv) {
            const __bf16* qp = Qb + base + (size_t)q * DQ;
            const __bf16* kp = Kb + base + (size_t)(q + 1 + lane) * DQ;
            float acc = 0.0f;
#pragma unroll
            for (int c = 0; c < 8; ++c) {
                union { uint4 qq; __bf16 e[8]; } uq, uk;
                uq.qq = *(const uint4*)(qp + c * 8);
                uk.qq = *(const uint4*)(kp + c * 8);
#pragma unroll
                for (int e = 0; e < 8; ++e) acc += (float)uq.e[e] * (float)uk.e[e];
            }
            s = acc * 0.125f;
        }
        float m = s;
#pragma unroll
        for (int off = 16; off; off >>= 1) m = fmaxf(m, __shfl_xor(m, off, 32));
        float p = __expf(s - m);
        float sum = p;
#pragma unroll
        for (int off = 16; off; off >>= 1) sum += __shfl_xor(sum, off, 32);
        p /= sum;
        for (int j = 0; j < nkv; ++j) {
            const float pj = __shfl(p, j, 32);
            union { unsigned u; __bf16 e[2]; } uv;
            uv.u = *(const unsigned*)(Vb + base + (size_t)(q + 1 + j) * DQ + d0);
            o0 += pj * (float)uv.e[0];
            o1 += pj * (float)uv.e[1];
        }
    }
    const size_t rr = (size_t)b * SQ + q;
    bwdF[rr * DQ + h * HDQ + d0]     = o0;
    bwdF[rr * DQ + h * HDQ + d0 + 1] = o1;
    comb[rr * (2 * DQ) + DQ + h * HDQ + d0]     = (__bf16)o0;
    comb[rr * (2 * DQ) + DQ + h * HDQ + d0 + 1] = (__bf16)o1;
}

// ---------------------------------------------------------------------------
// block-wide sum over 256 threads
// ---------------------------------------------------------------------------
__device__ inline float block_sum256(float v, float* sh) {
    const int lane = threadIdx.x & 31, w = threadIdx.x >> 5;
#pragma unroll
    for (int o = 16; o; o >>= 1) v += __shfl_xor(v, o, 32);
    if (lane == 0) sh[w] = v;
    __syncthreads();
    float r = (lane < 8) ? sh[lane] : 0.0f;
    if (w == 0) {
#pragma unroll
        for (int o = 4; o; o >>= 1) r += __shfl_xor(r, o, 32);
        if (lane == 0) sh[0] = r;
    }
    __syncthreads();
    const float out = sh[0];
    __syncthreads();
    return out;
}

// gate = sigmoid(LN(glin; gln)); fused = fwd + sigmoid(bstr)*0.3*gate*bwd
__global__ __launch_bounds__(256) void gate_fuse(
    const float* __restrict__ glin, const float* __restrict__ gln_g,
    const float* __restrict__ gln_b, const float* __restrict__ bstr,
    const float* __restrict__ fwdF, const float* __restrict__ bwdF,
    __bf16* __restrict__ fusedb)
{
    __shared__ float sh[8];
    const int row = blockIdx.x, tid = threadIdx.x;
    const float* g = glin + (size_t)row * DQ;
    float x[4];
#pragma unroll
    for (int i = 0; i < 4; ++i) x[i] = g[tid + i * 256];
    const float mean = block_sum256(x[0] + x[1] + x[2] + x[3], sh) * (1.0f / DQ);
    float vs = 0.0f;
#pragma unroll
    for (int i = 0; i < 4; ++i) { const float d = x[i] - mean; vs += d * d; }
    const float var  = block_sum256(vs, sh) * (1.0f / DQ);
    const float rstd = rsqrtf(var + LN_EPS);
    const float strength = 0.3f / (1.0f + __expf(-bstr[0]));
#pragma unroll
    for (int i = 0; i < 4; ++i) {
        const int idx = tid + i * 256;
        const float ln = (x[i] - mean) * rstd * gln_g[idx] + gln_b[idx];
        const float gate = 1.0f / (1.0f + __expf(-ln));
        const float f = fwdF[(size_t)row * DQ + idx] +
                        strength * gate * bwdF[(size_t)row * DQ + idx];
        fusedb[(size_t)row * DQ + idx] = (__bf16)f;
    }
}

// out = LN(x + y; ln)
__global__ __launch_bounds__(256) void final_ln(
    const float* __restrict__ x, const float* __restrict__ y,
    const float* __restrict__ ln_g, const float* __restrict__ ln_b,
    float* __restrict__ out)
{
    __shared__ float sh[8];
    const int row = blockIdx.x, tid = threadIdx.x;
    float t[4];
#pragma unroll
    for (int i = 0; i < 4; ++i) {
        const size_t idx = (size_t)row * DQ + tid + i * 256;
        t[i] = x[idx] + y[idx];
    }
    const float mean = block_sum256(t[0] + t[1] + t[2] + t[3], sh) * (1.0f / DQ);
    float vs = 0.0f;
#pragma unroll
    for (int i = 0; i < 4; ++i) { const float d = t[i] - mean; vs += d * d; }
    const float var  = block_sum256(vs, sh) * (1.0f / DQ);
    const float rstd = rsqrtf(var + LN_EPS);
#pragma unroll
    for (int i = 0; i < 4; ++i) {
        const int c = tid + i * 256;
        out[(size_t)row * DQ + c] = (t[i] - mean) * rstd * ln_g[c] + ln_b[c];
    }
}

// ---------------------------------------------------------------------------
// Host orchestration
// ---------------------------------------------------------------------------
extern "C" void kernel_launch(void* const* d_in, const int* in_sizes, int n_in,
                              void* d_out, int out_size, void* d_ws, size_t ws_size,
                              hipStream_t stream)
{
    (void)in_sizes; (void)n_in; (void)out_size; (void)ws_size;
    const float* x      = (const float*)d_in[0];
    const float* fq_w   = (const float*)d_in[1];  const float* fq_b = (const float*)d_in[2];
    const float* fk_w   = (const float*)d_in[3];  const float* fk_b = (const float*)d_in[4];
    const float* fv_w   = (const float*)d_in[5];  const float* fv_b = (const float*)d_in[6];
    const float* bq_w   = (const float*)d_in[7];  const float* bq_b = (const float*)d_in[8];
    const float* bk_w   = (const float*)d_in[9];  const float* bk_b = (const float*)d_in[10];
    const float* bv_w   = (const float*)d_in[11]; const float* bv_b = (const float*)d_in[12];
    const float* gate_w = (const float*)d_in[13]; const float* gate_b = (const float*)d_in[14];
    const float* gln_g  = (const float*)d_in[15]; const float* gln_b  = (const float*)d_in[16];
    const float* bstr   = (const float*)d_in[17];
    const float* out_w  = (const float*)d_in[18]; const float* out_b  = (const float*)d_in[19];
    const float* ln_g   = (const float*)d_in[20]; const float* ln_b   = (const float*)d_in[21];

    char* ws = (char*)d_ws;
    size_t off = 0;
    auto alloc = [&](size_t bytes) -> void* {
        off = (off + 255) & ~(size_t)255;
        void* p = ws + off;
        off += bytes;
        return p;
    };

    __bf16* xb      = (__bf16*)alloc((size_t)MQ * DQ * 2);
    __bf16* wt_fq   = (__bf16*)alloc((size_t)DQ * DQ * 2);     // [N,K]
    __bf16* wt_fk   = (__bf16*)alloc((size_t)DQ * DQ * 2);
    __bf16* wt_fv   = (__bf16*)alloc((size_t)DQ * DQ * 2);
    __bf16* wt_bq   = (__bf16*)alloc((size_t)DQ * DQ * 2);
    __bf16* wt_bk   = (__bf16*)alloc((size_t)DQ * DQ * 2);
    __bf16* wt_bv   = (__bf16*)alloc((size_t)DQ * DQ * 2);
    __bf16* wt_gate = (__bf16*)alloc((size_t)2 * DQ * DQ * 2); // [1024, 2048]
    __bf16* wt_out  = (__bf16*)alloc((size_t)DQ * DQ * 2);
    __bf16* Qf      = (__bf16*)alloc((size_t)MQ * DQ * 2);
    __bf16* Kf      = (__bf16*)alloc((size_t)MQ * DQ * 2);
    __bf16* Vf      = (__bf16*)alloc((size_t)MQ * DQ * 2);
    __bf16* Vft     = (__bf16*)alloc((size_t)MQ * DQ * 2);     // [B*H, HD, S]
    __bf16* Qw      = (__bf16*)alloc((size_t)MQ * DQ * 2);
    __bf16* Kw      = (__bf16*)alloc((size_t)MQ * DQ * 2);
    __bf16* Vw      = (__bf16*)alloc((size_t)MQ * DQ * 2);
    float*  fwdF    = (float*)alloc((size_t)MQ * DQ * 4);
    float*  bwdF    = (float*)alloc((size_t)MQ * DQ * 4);
    __bf16* comb    = (__bf16*)alloc((size_t)MQ * 2 * DQ * 2);
    float*  glin    = (float*)alloc((size_t)MQ * DQ * 4);
    __bf16* fusedb  = (__bf16*)alloc((size_t)MQ * DQ * 2);
    float*  yout    = (float*)alloc((size_t)MQ * DQ * 4);

    // --- conversions: x straight, weights transposed to [N,K] ---
    k_cvt_bf16<<<(MQ * DQ + 255) / 256, 256, 0, stream>>>(x, xb, MQ * DQ);
    auto cvtT = [&](const float* src, __bf16* dst, int R, int C) {
        k_cvt_T<<<dim3(C / 32, R / 32), 256, 0, stream>>>(src, dst, R, C);
    };
    cvtT(fq_w, wt_fq, DQ, DQ); cvtT(fk_w, wt_fk, DQ, DQ); cvtT(fv_w, wt_fv, DQ, DQ);
    cvtT(bq_w, wt_bq, DQ, DQ); cvtT(bk_w, wt_bk, DQ, DQ); cvtT(bv_w, wt_bv, DQ, DQ);
    cvtT(gate_w, wt_gate, 2 * DQ, DQ);
    cvtT(out_w, wt_out, DQ, DQ);

    // --- QKV projections (bf16 out) ---
    dim3 gproj(DQ / 128, MQ / 128);
    gemm_wmma<__bf16><<<gproj, 256, 0, stream>>>(xb, wt_fq, fq_b, Qf, MQ, DQ, DQ);
    gemm_wmma<__bf16><<<gproj, 256, 0, stream>>>(xb, wt_fk, fk_b, Kf, MQ, DQ, DQ);
    gemm_wmma<__bf16><<<gproj, 256, 0, stream>>>(xb, wt_fv, fv_b, Vf, MQ, DQ, DQ);
    gemm_wmma<__bf16><<<gproj, 256, 0, stream>>>(xb, wt_bq, bq_b, Qw, MQ, DQ, DQ);
    gemm_wmma<__bf16><<<gproj, 256, 0, stream>>>(xb, wt_bk, bk_b, Kw, MQ, DQ, DQ);
    gemm_wmma<__bf16><<<gproj, 256, 0, stream>>>(xb, wt_bv, bv_b, Vw, MQ, DQ, DQ);

    // --- per-head transpose of forward V ---
    k_transpose_v<<<dim3(SQ / 32, BQ * HQ), 256, 0, stream>>>(Vf, Vft);

    // --- attentions ---
    fwd_attn<<<dim3(SQ / 128, BQ * HQ), 256, 0, stream>>>(Qf, Kf, Vft, fwdF, comb);
    bwd_attn<<<(BQ * HQ * SQ) / 8, 256, 0, stream>>>(Qw, Kw, Vw, bwdF, comb);

    // --- gate GEMM (K=2048, f32 out), LN+sigmoid+fuse ---
    gemm_wmma<float><<<gproj, 256, 0, stream>>>(comb, wt_gate, gate_b, glin, MQ, DQ, 2 * DQ);
    gate_fuse<<<MQ, 256, 0, stream>>>(glin, gln_g, gln_b, bstr, fwdF, bwdF, fusedb);

    // --- output projection + residual LN ---
    gemm_wmma<float><<<gproj, 256, 0, stream>>>(fusedb, wt_out, out_b, yout, MQ, DQ, DQ);
    final_ln<<<MQ, 256, 0, stream>>>(x, yout, ln_g, ln_b, (float*)d_out);
}
